// MultiHeadAttention_8504035246474
// MI455X (gfx1250) — compile-verified, matrix operand order checked by reading, not run on hardware
//
#include <hip/hip_runtime.h>
#include <hip/hip_bf16.h>

// ---------------------------------------------------------------------------
// MHA forward for MI455X (gfx1250, wave32, WMMA 16x16x32 f16 -> f32 acc).
// B=4, S=2048, D=512, H=8, DK=DV=64.
// Outputs (concatenated in d_out): output [B,S,512] f32, attn [B,H,S,S] f32.
// Register blocking: each wave computes a 16x64 output strip with 4
// independent accumulators (kills the WMMA->WMMA RAW hazard NOPs and reuses
// the A fragment 4x).
// ---------------------------------------------------------------------------

#define BB   4
#define SS   2048
#define DD   512
#define HH   8
#define DKK  64

typedef __attribute__((ext_vector_type(16))) _Float16 v16h;
typedef __attribute__((ext_vector_type(8)))  float    v8f;

union Frag16 {
  v16h     v;
  _Float16 h[16];
  unsigned u[8];
};

// A/B fragment loader from a row-major f16 matrix where the K dimension is
// contiguous (ld = leading dimension in elements). CDNA5 16-bit 16x32 operand
// layout: lanes 0-15 hold row/col = lane with K pairs
// {0,1},{2,3},{4,5},{6,7},{16,17},{18,19},{20,21},{22,23} in VGPRs 0..7;
// lanes 16-31 hold the +8 K-shifted pairs.
__device__ inline v16h load_frag_f16(const _Float16* __restrict__ base, int ld) {
  const int lane = threadIdx.x & 31;
  const int half = lane >> 4;
  const int l    = lane & 15;
  const _Float16* p = base + (long)l * ld + half * 8;
  Frag16 f;
#pragma unroll
  for (int j = 0; j < 8; ++j) {
    const int k = (j < 4) ? (2 * j) : (16 + 2 * (j - 4));
    f.u[j] = *(const unsigned*)(p + k);   // packed pair (k, k+1)
  }
  return f.v;
}

// Same layout, but the source matrix is f32; convert to f16 on load.
__device__ inline v16h load_frag_f32(const float* __restrict__ base, int ld) {
  const int lane = threadIdx.x & 31;
  const int half = lane >> 4;
  const int l    = lane & 15;
  const float* p = base + (long)l * ld + half * 8;
  Frag16 f;
#pragma unroll
  for (int j = 0; j < 8; ++j) {
    const int k = (j < 4) ? (2 * j) : (16 + 2 * (j - 4));
    f.h[2 * j]     = (_Float16)p[k];
    f.h[2 * j + 1] = (_Float16)p[k + 1];
  }
  return f.v;
}

__device__ inline v8f wmma16(v16h a, v16h b, v8f c) {
  // (neg_a, A, neg_b, B, c_mod, C, reuse_a, reuse_b)
  return __builtin_amdgcn_wmma_f32_16x16x32_f16(false, a, false, b, (short)0, c,
                                                false, false);
}

// ---------------------------------------------------------------------------
// K0: convert the four weight matrices (each 512x512 f32) to f16 in workspace.
// ---------------------------------------------------------------------------
__global__ __launch_bounds__(256) void mha_cvtw_kernel(
    const float* __restrict__ wq, const float* __restrict__ wk,
    const float* __restrict__ wv, const float* __restrict__ wfc,
    _Float16* __restrict__ o) {
  const int i = blockIdx.x * 256 + threadIdx.x;   // 0 .. 4*262144-1
  const int which = i >> 18;
  const int j = i & 262143;
  const float* src = (which == 0) ? wq : (which == 1) ? wk : (which == 2) ? wv : wfc;
  o[i] = (_Float16)src[j];
}

// ---------------------------------------------------------------------------
// K1: Q/K/V projections:  P = X @ W^T   (M=B*S=8192, N=512, K=512)
// One block per M-tile; wave w owns N-tiles [4w, 4w+4) -> full N row/block.
// Q,K stored [b][h][s][dk] f16; V stored transposed [b][h][dv][s] f16.
// ---------------------------------------------------------------------------
__global__ __launch_bounds__(256) void mha_proj_kernel(
    const float* __restrict__ xq, const float* __restrict__ xk,
    const float* __restrict__ xv, const _Float16* __restrict__ wq,
    const _Float16* __restrict__ wk, const _Float16* __restrict__ wv,
    _Float16* __restrict__ qo, _Float16* __restrict__ ko,
    _Float16* __restrict__ vto) {
  const int wave = threadIdx.x >> 5;
  const int lane = threadIdx.x & 31;
  const int half = lane >> 4;
  const int l    = lane & 15;
  const int mtile = blockIdx.x;              // 0..511
  const int which = blockIdx.z;              // 0=Q 1=K 2=V

  const float*    X = (which == 0) ? xq : (which == 1) ? xk : xv;
  const _Float16* W = (which == 0) ? wq : (which == 1) ? wk : wv;

  const float*    Ab = X + (long)mtile * 16 * DD;
  const _Float16* Bb = W + (long)wave * 64 * DD;   // 4 n-tiles per wave

  v8f zero = {};
  v8f acc[4] = {zero, zero, zero, zero};
  for (int k0 = 0; k0 < DD; k0 += 32) {
    v16h a = load_frag_f32(Ab + k0, DD);
#pragma unroll
    for (int t = 0; t < 4; ++t) {
      v16h b = load_frag_f16(Bb + (long)t * 16 * DD + k0, DD);
      acc[t] = wmma16(a, b, acc[t]);
    }
  }

  const int gm0 = mtile * 16;      // 2048 % 16 == 0 -> tile within one batch
  const int b0  = gm0 >> 11;
  const int s0  = gm0 & 2047;
#pragma unroll
  for (int t = 0; t < 4; ++t) {
    const int n  = wave * 64 + t * 16 + l;
    const int h  = n >> 6;
    const int dk = n & 63;
#pragma unroll
    for (int r = 0; r < 8; ++r) {
      const int s = s0 + r + half * 8;
      const _Float16 val = (_Float16)acc[t][r];
      if (which == 2) {
        vto[((long)(b0 * HH + h) * DKK + dk) * SS + s] = val;
      } else {
        _Float16* dst = (which == 0) ? qo : ko;
        dst[((long)(b0 * HH + h) * SS + s) * DKK + dk] = val;
      }
    }
  }
}

// ---------------------------------------------------------------------------
// K2: scores = (Q @ K^T) * 0.125 + other, masked -> f32 into the attn region
// of d_out (softmaxed in place next). Wave computes a 16x64 strip.
// ---------------------------------------------------------------------------
__global__ __launch_bounds__(256) void mha_qk_kernel(
    const _Float16* __restrict__ qm, const _Float16* __restrict__ km,
    const float* __restrict__ other, const int* __restrict__ mask,
    float* __restrict__ attn) {
  const int wave = threadIdx.x >> 5;
  const int lane = threadIdx.x & 31;
  const int half = lane >> 4;
  const int l    = lane & 15;
  const int ng    = blockIdx.x * 8 + wave;   // key tile-group 0..31 (64 keys)
  const int mtile = blockIdx.y;              // query tile 0..127
  const int bh    = blockIdx.z;              // 0..31
  const int b     = bh >> 3;

  const _Float16* Qb = qm + (long)bh * SS * DKK + (long)mtile * 16 * DKK;
  const _Float16* Kb = km + (long)bh * SS * DKK + (long)ng * 64 * DKK;

  v8f zero = {};
  v8f acc[4] = {zero, zero, zero, zero};
#pragma unroll
  for (int k0 = 0; k0 < DKK; k0 += 32) {
    v16h a = load_frag_f16(Qb + k0, DKK);
#pragma unroll
    for (int t = 0; t < 4; ++t) {
      v16h bf = load_frag_f16(Kb + (long)t * 16 * DKK + k0, DKK);
      acc[t] = wmma16(a, bf, acc[t]);
    }
  }

  const int q0 = mtile * 16;
  float*       outp = attn  + ((long)bh * SS + q0) * SS + ng * 64 + l;
  const float* othp = other + ((long)b  * SS + q0) * SS + ng * 64 + l;
  const int*   mskp = mask  + ((long)b  * SS + q0) * SS + ng * 64 + l;
#pragma unroll
  for (int t = 0; t < 4; ++t) {
#pragma unroll
    for (int r = 0; r < 8; ++r) {
      const long off = (long)(r + half * 8) * SS + t * 16;
      float v = acc[t][r] * 0.125f + othp[off];
      if (mskp[off] != 0) v = -1e9f;
      outp[off] = v;
    }
  }
}

// ---------------------------------------------------------------------------
// K3: in-place row softmax over the attn region (rows of length S=2048).
// One 256-thread block per (b,h,q) row; 8 elements per thread.
// ---------------------------------------------------------------------------
__global__ __launch_bounds__(256) void mha_softmax_kernel(float* __restrict__ attn) {
  const long row = blockIdx.x;
  float* p = attn + row * (long)SS;
  __shared__ float red[256];

  float loc[8];
  float m = -3.4e38f;
#pragma unroll
  for (int i = 0; i < 8; ++i) {
    loc[i] = p[threadIdx.x + 256 * i];
    m = fmaxf(m, loc[i]);
  }
  red[threadIdx.x] = m;
  __syncthreads();
  for (int s = 128; s > 0; s >>= 1) {
    if (threadIdx.x < (unsigned)s)
      red[threadIdx.x] = fmaxf(red[threadIdx.x], red[threadIdx.x + s]);
    __syncthreads();
  }
  m = red[0];
  __syncthreads();

  float sum = 0.f;
#pragma unroll
  for (int i = 0; i < 8; ++i) {
    loc[i] = __expf(loc[i] - m);
    sum += loc[i];
  }
  red[threadIdx.x] = sum;
  __syncthreads();
  for (int s = 128; s > 0; s >>= 1) {
    if (threadIdx.x < (unsigned)s)
      red[threadIdx.x] += red[threadIdx.x + s];
    __syncthreads();
  }
  const float inv = 1.0f / red[0];
#pragma unroll
  for (int i = 0; i < 8; ++i) p[threadIdx.x + 256 * i] = loc[i] * inv;
}

// ---------------------------------------------------------------------------
// K4: context = attn @ V.  One wave owns one q-tile and ALL 4 dv tiles: the
// f32->f16 converted attn fragment is loaded once per k-step and reused 4x.
// B operand = V^T f16 [b][h][dv][s] (K pairs contiguous). Output context f16
// [B*S, H*DV] row-major for the final projection.
// ---------------------------------------------------------------------------
__global__ __launch_bounds__(256) void mha_pv_kernel(
    const float* __restrict__ attn, const _Float16* __restrict__ vt,
    _Float16* __restrict__ ctx) {
  const int wave = threadIdx.x >> 5;
  const int lane = threadIdx.x & 31;
  const int half = lane >> 4;
  const int l    = lane & 15;
  const int mtile = blockIdx.x * 8 + wave;  // query tile 0..127
  const int bh    = blockIdx.y;             // 0..31
  const int b     = bh >> 3;
  const int h     = bh & 7;

  const float*    Ab = attn + ((long)bh * SS + (long)mtile * 16) * SS;
  const _Float16* Bb = vt   + (long)bh * DKK * SS;   // dv tiles at +t*16*SS

  v8f zero = {};
  v8f acc[4] = {zero, zero, zero, zero};
  for (int k0 = 0; k0 < SS; k0 += 32) {
    __builtin_prefetch(Ab + k0 + 32, 0, 0);
    v16h a = load_frag_f32(Ab + k0, SS);
#pragma unroll
    for (int t = 0; t < 4; ++t) {
      v16h bf = load_frag_f16(Bb + (long)t * 16 * SS + k0, SS);
      acc[t] = wmma16(a, bf, acc[t]);
    }
  }

  _Float16* out = ctx + ((long)b * SS + mtile * 16) * DD + h * DKK + l;
#pragma unroll
  for (int t = 0; t < 4; ++t) {
#pragma unroll
    for (int r = 0; r < 8; ++r)
      out[(long)(r + half * 8) * DD + t * 16] = (_Float16)acc[t][r];
  }
}

// ---------------------------------------------------------------------------
// K5: output = context @ W_fc^T  (M=8192, N=512, K=512), f32 result.
// One block per M-tile; wave w owns N-tiles [4w, 4w+4).
// ---------------------------------------------------------------------------
__global__ __launch_bounds__(256) void mha_out_kernel(
    const _Float16* __restrict__ ctx, const _Float16* __restrict__ wfc,
    float* __restrict__ out) {
  const int wave = threadIdx.x >> 5;
  const int lane = threadIdx.x & 31;
  const int half = lane >> 4;
  const int l    = lane & 15;
  const int mtile = blockIdx.x;              // 0..511

  const _Float16* Ab = ctx + (long)mtile * 16 * DD;
  const _Float16* Bb = wfc + (long)wave * 64 * DD;

  v8f zero = {};
  v8f acc[4] = {zero, zero, zero, zero};
  for (int k0 = 0; k0 < DD; k0 += 32) {
    v16h a = load_frag_f16(Ab + k0, DD);
#pragma unroll
    for (int t = 0; t < 4; ++t) {
      v16h b = load_frag_f16(Bb + (long)t * 16 * DD + k0, DD);
      acc[t] = wmma16(a, b, acc[t]);
    }
  }

  float* o = out + ((long)mtile * 16) * DD + wave * 64 + l;
#pragma unroll
  for (int t = 0; t < 4; ++t) {
#pragma unroll
    for (int r = 0; r < 8; ++r)
      o[(long)(r + half * 8) * DD + t * 16] = acc[t][r];
  }
}

// ---------------------------------------------------------------------------
// Host-side launch. Workspace layout (f16 elements):
//   [0)        wq16  262144
//   [262144)   wk16  262144
//   [524288)   wv16  262144
//   [786432)   wfc16 262144
//   [1048576)  q_ws   4194304   ([b][h][s][dk])
//   [+]        k_ws   4194304   ([b][h][s][dk])
//   [+]        vt_ws  4194304   ([b][h][dv][s])
//   [+]        ctx_ws 4194304   ([b*s][h*dv])
// total = 17,825,792 f16 = ~35.7 MB.
// Scores/attn live in d_out's attn region (written once, softmaxed in place).
// ---------------------------------------------------------------------------
extern "C" void kernel_launch(void* const* d_in, const int* in_sizes, int n_in,
                              void* d_out, int out_size, void* d_ws, size_t ws_size,
                              hipStream_t stream) {
  const float* input_Q   = (const float*)d_in[0];
  const float* input_K   = (const float*)d_in[1];
  const float* input_V   = (const float*)d_in[2];
  const int*   attn_mask = (const int*)d_in[3];   // bool mask, int32 per harness
  const float* other     = (const float*)d_in[4];
  const float* W_Q       = (const float*)d_in[5];
  const float* W_K       = (const float*)d_in[6];
  const float* W_V       = (const float*)d_in[7];
  const float* W_fc      = (const float*)d_in[8];

  float* out_proj = (float*)d_out;                       // [B,S,512]
  float* attn_out = out_proj + (long)BB * SS * DD;       // [B,H,S,S]

  _Float16* p      = (_Float16*)d_ws;
  _Float16* wq16   = p;                 // 512*512
  _Float16* wk16   = wq16 + 262144;
  _Float16* wv16   = wk16 + 262144;
  _Float16* wfc16  = wv16 + 262144;
  _Float16* q_ws   = wfc16 + 262144;
  _Float16* k_ws   = q_ws + 4194304;
  _Float16* vt_ws  = k_ws + 4194304;
  _Float16* ctx_ws = vt_ws + 4194304;

  // K0: weights f32 -> f16
  mha_cvtw_kernel<<<dim3(4096), dim3(256), 0, stream>>>(W_Q, W_K, W_V, W_fc, wq16);

  // K1: Q/K/V projections (z = which projection)
  mha_proj_kernel<<<dim3(512, 1, 3), dim3(256), 0, stream>>>(
      input_Q, input_K, input_V, wq16, wk16, wv16, q_ws, k_ws, vt_ws);

  // K2: scores + scale + bias + mask -> attn region of d_out
  mha_qk_kernel<<<dim3(4, 128, 32), dim3(256), 0, stream>>>(
      q_ws, k_ws, other, attn_mask, attn_out);

  // K3: softmax in place (one block per (b,h,q) row)
  mha_softmax_kernel<<<dim3(BB * HH * SS), dim3(256), 0, stream>>>(attn_out);

  // K4: context = attn @ V (one wave: 1 q-tile x all 4 dv tiles)
  mha_pv_kernel<<<dim3(16, 32), dim3(256), 0, stream>>>(attn_out, vt_ws, ctx_ws);

  // K5: output = context @ W_fc^T
  mha_out_kernel<<<dim3(512), dim3(256), 0, stream>>>(ctx_ws, wfc16, out_proj);
}
